// AutoformerAttention_53257594470782
// MI455X (gfx1250) — matrix operand order repaired and checked by reading, not compiled, not run on hardware
//
#include <hip/hip_runtime.h>
#include <hip/hip_bf16.h>

typedef __attribute__((ext_vector_type(16))) _Float16 v16h;
typedef __attribute__((ext_vector_type(8)))  _Float16 v8h;
typedef __attribute__((ext_vector_type(8)))  float    v8f;

#define Dm 512
#define Lm 3072
#define Bm 8
#define KSPAD 12
#define NCH 12
#define CH  256
#define TOPK 40
#define AGG_T 8

// ---------------- LayerNorm: one block per (b,l) row ----------------
__global__ __launch_bounds__(256) void ln_kernel(const float* __restrict__ x,
                                                 const float* __restrict__ gamma,
                                                 const float* __restrict__ beta,
                                                 float* __restrict__ xn) {
  int row = blockIdx.x;               // b*L + l
  int tid = threadIdx.x;              // 256
  const float* xr = x + (size_t)row * Dm;
  float v0 = xr[tid], v1 = xr[tid + 256];
  __shared__ float red[256];
  red[tid] = v0 + v1;
  __syncthreads();
  for (int s = 128; s > 0; s >>= 1) {
    if (tid < s) red[tid] += red[tid + s];
    __syncthreads();
  }
  float mu = red[0] * (1.0f / 512.0f);
  __syncthreads();
  float d0 = v0 - mu, d1 = v1 - mu;
  red[tid] = d0 * d0 + d1 * d1;
  __syncthreads();
  for (int s = 128; s > 0; s >>= 1) {
    if (tid < s) red[tid] += red[tid + s];
    __syncthreads();
  }
  float var = red[0] * (1.0f / 512.0f);
  float rs = rsqrtf(var + 1e-5f);
  float* o = xn + (size_t)row * Dm;
  o[tid]       = d0 * rs * gamma[tid]       + beta[tid];
  o[tid + 256] = d1 * rs * gamma[tid + 256] + beta[tid + 256];
}

// ------- series decomp: moving average (k=25, count_include_pad=False) -------
__global__ __launch_bounds__(256) void decomp_kernel(const float* __restrict__ xn,
                                                     _Float16* __restrict__ sea,
                                                     float* __restrict__ pssum) {
  int blk = blockIdx.x;               // B * 2 * NCH = 192 blocks
  int b = blk / (2 * NCH);
  int r = blk % (2 * NCH);
  int colhalf = r / NCH;
  int chunk = r % NCH;
  int d = colhalf * 256 + threadIdx.x;
  int t0 = chunk * CH;

  const float* col = xn + (size_t)b * Lm * Dm + d;
  float wsum = 0.0f;
  int lo0 = t0 - KSPAD; if (lo0 < 0) lo0 = 0;
  int hi0 = t0 + KSPAD; if (hi0 > Lm - 1) hi0 = Lm - 1;
  for (int l = lo0; l <= hi0; ++l) wsum += col[(size_t)l * Dm];

  float ps = 0.0f;
  for (int t = t0; t < t0 + CH; ++t) {
    int lo = t - KSPAD; if (lo < 0) lo = 0;
    int hi = t + KSPAD; if (hi > Lm - 1) hi = Lm - 1;
    float cnt = (float)(hi - lo + 1);
    float xv = col[(size_t)t * Dm];
    float s = xv - wsum / cnt;
    sea[(size_t)b * Lm * Dm + (size_t)t * Dm + d] = (_Float16)s;
    ps += s;
    int addl = t + KSPAD + 1;
    if (addl <= Lm - 1) wsum += col[(size_t)addl * Dm];
    int reml = t - KSPAD;
    if (reml >= 0) wsum -= col[(size_t)reml * Dm];
  }
  pssum[((size_t)b * NCH + chunk) * Dm + d] = ps;
}

// -------- convert + transpose a 512x512 f32 weight to f16 (Wt[n][k]) --------
__global__ __launch_bounds__(256) void wcvt_kernel(const float* __restrict__ W,
                                                   _Float16* __restrict__ Wt) {
  int k = blockIdx.x;
  for (int n = threadIdx.x; n < Dm; n += 256)
    Wt[(size_t)n * Dm + k] = (_Float16)W[(size_t)k * Dm + n];
}

// ------- Q/K column-sum statistics -> mean_value[b, c] (c in [0,64)) -------
__global__ __launch_bounds__(512) void qk_stats_kernel(const float* __restrict__ pssum,
                                                       const float* __restrict__ Wq,
                                                       const float* __restrict__ bq,
                                                       const float* __restrict__ Wk,
                                                       const float* __restrict__ bk,
                                                       float* __restrict__ mv) {
  int b = blockIdx.x;
  int d = threadIdx.x;                // 512 threads
  __shared__ float ss[512], qs[512], ks[512];
  float s = 0.0f;
  for (int c = 0; c < NCH; ++c) s += pssum[((size_t)b * NCH + c) * Dm + d];
  ss[d] = s;
  __syncthreads();
  float qc = 3072.0f * bq[d], kc = 3072.0f * bk[d];
  for (int j = 0; j < Dm; ++j) {
    float sj = ss[j];
    qc = fmaf(sj, Wq[(size_t)j * Dm + d], qc);
    kc = fmaf(sj, Wk[(size_t)j * Dm + d], kc);
  }
  qs[d] = qc; ks[d] = kc;
  __syncthreads();
  if (d < 64) {
    float acc = 0.0f;
    for (int h = 0; h < 8; ++h) acc += qs[h * 64 + d] * ks[h * 64 + d];
    mv[b * 64 + d] = acc * (1.0f / (8.0f * 3072.0f));
  }
}

// ---- top-40 over batch-mean, per-batch softmax, scatter to 64-tap filter ----
__global__ void topk_kernel(const float* __restrict__ mv, float* __restrict__ cw) {
  __shared__ float gm[64];
  __shared__ int idx[TOPK];
  int tid = threadIdx.x;              // 64 threads
  if (tid < 64) {
    float s = 0.0f;
    for (int b = 0; b < Bm; ++b) s += mv[b * 64 + tid];
    gm[tid] = s * 0.125f;
  }
  __syncthreads();
  if (tid == 0) {
    bool taken[64];
    for (int i = 0; i < 64; ++i) taken[i] = false;
    for (int k = 0; k < TOPK; ++k) {
      int best = 0; float bvv = -3.4e38f;
      for (int c = 0; c < 64; ++c)
        if (!taken[c] && gm[c] > bvv) { bvv = gm[c]; best = c; }
      taken[best] = true;
      idx[k] = best;
    }
  }
  __syncthreads();
  if (tid < 64)
    for (int b = 0; b < Bm; ++b) cw[b * 64 + tid] = 0.0f;
  __syncthreads();
  if (tid < Bm) {
    int b = tid;
    float vals[TOPK];
    float m = -3.4e38f;
    for (int k = 0; k < TOPK; ++k) {
      vals[k] = mv[b * 64 + idx[k]];
      m = fmaxf(m, vals[k]);
    }
    float ssum = 0.0f;
    for (int k = 0; k < TOPK; ++k) { vals[k] = __expf(vals[k] - m); ssum += vals[k]; }
    float inv = 1.0f / ssum;
    for (int k = 0; k < TOPK; ++k) cw[b * 64 + idx[k]] = vals[k] * inv;
  }
}

// -------- delay aggregation: agg[b,t,d] = sum_s cw[b,s] * V[b,(t+s)%L,d] -----
// Each thread produces AGG_T consecutive t outputs while streaming V rows once:
// ~ (AGG_T+63)/AGG_T loads per output instead of 64.
__global__ __launch_bounds__(256) void agg_kernel(const float* __restrict__ V,
                                                  const float* __restrict__ cw,
                                                  _Float16* __restrict__ agg) {
  int blk = blockIdx.x;               // B * (Lm/AGG_T) * 2 = 6144
  int b = blk / ((Lm / AGG_T) * 2);
  int r = blk % ((Lm / AGG_T) * 2);
  int tgrp = r >> 1;
  int half = r & 1;
  int t0 = tgrp * AGG_T;
  int d = half * 256 + threadIdx.x;
  __shared__ float w[64];
  if (threadIdx.x < 64) w[threadIdx.x] = cw[b * 64 + threadIdx.x];
  __syncthreads();
  const float* Vb = V + (size_t)b * Lm * Dm;

  float acc[AGG_T];
#pragma unroll
  for (int i = 0; i < AGG_T; ++i) acc[i] = 0.0f;

  for (int l = t0; l < t0 + AGG_T + 63; ++l) {
    int lmod = (l >= Lm) ? (l - Lm) : l;
    float v = Vb[(size_t)lmod * Dm + d];
#pragma unroll
    for (int i = 0; i < AGG_T; ++i) {
      int s = l - t0 - i;
      if (s >= 0 && s < 64) acc[i] = fmaf(w[s], v, acc[i]);  // w[s] uniform -> LDS broadcast
    }
  }
#pragma unroll
  for (int i = 0; i < AGG_T; ++i)
    agg[(size_t)b * Lm * Dm + (size_t)(t0 + i) * Dm + d] = (_Float16)acc[i];
}

// ------- WMMA GEMM: C[M,512] = A[M,512](f16) * Wt^T + bias (+resid) ---------
// Wt pre-transposed f16: Wt[n*512+k] = W[k,n]. One wave = 16x64 strip
// (1 M-tile x 4 N-tiles) so each A fragment feeds 4 WMMAs.
__global__ __launch_bounds__(256) void gemm_f16_kernel(const _Float16* __restrict__ A,
                                                       const _Float16* __restrict__ Wt,
                                                       const float* __restrict__ bias,
                                                       const float* __restrict__ resid,
                                                       float* __restrict__ C) {
  int wave = threadIdx.x >> 5;
  int lane = threadIdx.x & 31;
  int half = lane >> 4;
  int ml = lane & 15;
  int gtile = blockIdx.x * 8 + wave;  // 12288 wave-jobs
  int tm = gtile >> 3;                // 1536 M-tiles
  int ts = gtile & 7;                 // 8 N-strips of 64
  int m0 = tm * 16, n0 = ts * 64;

  const _Float16* arow = A + (size_t)(m0 + ml) * Dm;
  const _Float16* brow0 = Wt + (size_t)(n0 +  0 + ml) * Dm;
  const _Float16* brow1 = Wt + (size_t)(n0 + 16 + ml) * Dm;
  const _Float16* brow2 = Wt + (size_t)(n0 + 32 + ml) * Dm;
  const _Float16* brow3 = Wt + (size_t)(n0 + 48 + ml) * Dm;

  v8f acc0, acc1, acc2, acc3;
#pragma unroll
  for (int i = 0; i < 8; ++i) { acc0[i] = 0.0f; acc1[i] = 0.0f; acc2[i] = 0.0f; acc3[i] = 0.0f; }

#pragma unroll
  for (int k0 = 0; k0 < Dm; k0 += 32) {
    // A 16x32 f16 fragment: lane holds K = 16*half + 0..15
    v8h a0 = *(const v8h*)(arow + k0 + 16 * half);
    v8h a1 = *(const v8h*)(arow + k0 + 16 * half + 8);
    v16h a = __builtin_shufflevector(a0, a1, 0, 1, 2, 3, 4, 5, 6, 7,
                                             8, 9, 10, 11, 12, 13, 14, 15);
    // B 32x16 f16 fragments: lane holds K = 16*half + 0..15 of column (n0+j*16+ml)
    v16h b0 = *(const v16h*)(brow0 + k0 + 16 * half);
    v16h b1 = *(const v16h*)(brow1 + k0 + 16 * half);
    v16h b2 = *(const v16h*)(brow2 + k0 + 16 * half);
    v16h b3 = *(const v16h*)(brow3 + k0 + 16 * half);
    acc0 = __builtin_amdgcn_wmma_f32_16x16x32_f16(false, a, false, b0, (short)0, acc0, false, false);
    acc1 = __builtin_amdgcn_wmma_f32_16x16x32_f16(false, a, false, b1, (short)0, acc1, false, false);
    acc2 = __builtin_amdgcn_wmma_f32_16x16x32_f16(false, a, false, b2, (short)0, acc2, false, false);
    acc3 = __builtin_amdgcn_wmma_f32_16x16x32_f16(false, a, false, b3, (short)0, acc3, false, false);
  }

#pragma unroll
  for (int j = 0; j < 4; ++j) {
    v8f accj = (j == 0) ? acc0 : (j == 1) ? acc1 : (j == 2) ? acc2 : acc3;
    int col = n0 + j * 16 + ml;
    float bv = bias[col];
#pragma unroll
    for (int r = 0; r < 8; ++r) {
      int row = m0 + r + 8 * half;    // C layout: VGPR r -> M = r / r+8 by lane half
      size_t o = (size_t)row * Dm + col;
      float v = accj[r] + bv;
      if (resid) v += resid[o];
      C[o] = v;
    }
  }
}

extern "C" void kernel_launch(void* const* d_in, const int* in_sizes, int n_in,
                              void* d_out, int out_size, void* d_ws, size_t ws_size,
                              hipStream_t stream) {
  const float* x     = (const float*)d_in[0];
  const float* Wq    = (const float*)d_in[1];
  const float* Wk    = (const float*)d_in[2];
  const float* Wv    = (const float*)d_in[3];
  const float* Wo    = (const float*)d_in[4];
  const float* bq    = (const float*)d_in[5];
  const float* bk    = (const float*)d_in[6];
  const float* bv    = (const float*)d_in[7];
  const float* bo    = (const float*)d_in[8];
  const float* gamma = (const float*)d_in[9];
  const float* beta  = (const float*)d_in[10];

  const size_t N = (size_t)Bm * Lm * Dm;   // 12,582,912
  char* ws = (char*)d_ws;
  float*    xnV  = (float*)ws;                       // f32 [N]: xn, then V
  _Float16* seaA = (_Float16*)(ws + N * 4);          // f16 [N]: seasonal, then agg
  char* p = ws + N * 4 + N * 2;
  _Float16* WvT   = (_Float16*)p; p += (size_t)Dm * Dm * 2;
  _Float16* WoT   = (_Float16*)p; p += (size_t)Dm * Dm * 2;
  float*    pssum = (float*)p;    p += (size_t)Bm * NCH * Dm * 4;
  float*    mv    = (float*)p;    p += Bm * 64 * 4;
  float*    cw    = (float*)p;

  // 1. LayerNorm
  ln_kernel<<<Bm * Lm, 256, 0, stream>>>(x, gamma, beta, xnV);
  // 2. series decomposition -> seasonal (f16) + partial row sums
  decomp_kernel<<<Bm * 2 * NCH, 256, 0, stream>>>(xnV, seaA, pssum);
  // 3. weight convert+transpose to f16
  wcvt_kernel<<<Dm, 256, 0, stream>>>(Wv, WvT);
  wcvt_kernel<<<Dm, 256, 0, stream>>>(Wo, WoT);
  // 4. V = seasonal @ Wv + bv   (WMMA; writes over xn buffer)
  gemm_f16_kernel<<<(Bm * Lm / 16) * 8 / 8, 256, 0, stream>>>(seaA, WvT, bv, nullptr, xnV);
  // 5. mean_value[b,c] from column-sum identity (FFT eliminated)
  qk_stats_kernel<<<Bm, 512, 0, stream>>>(pssum, Wq, bq, Wk, bk, mv);
  // 6. top-40 + softmax + 64-tap per-batch filter
  topk_kernel<<<1, 64, 0, stream>>>(mv, cw);
  // 7. delay aggregation (writes agg f16 over seasonal buffer)
  agg_kernel<<<Bm * (Lm / AGG_T) * 2, 256, 0, stream>>>(xnV, cw, seaA);
  // 8. out = agg @ Wo + bo + x  ==  s2 + t2 (trend cancels algebraically)
  gemm_f16_kernel<<<(Bm * Lm / 16) * 8 / 8, 256, 0, stream>>>(seaA, WoT, bo, x, (float*)d_out);
}
